// LSTM_LRP_MultiLayer_67680094650709
// MI455X (gfx1250) — matrix operand order repaired and checked by reading, not compiled, not run on hardware
//
#include <hip/hip_runtime.h>
#include <stdint.h>

// ---------------------------------------------------------------------------
// Problem constants (from reference)
// ---------------------------------------------------------------------------
constexpr int B_  = 4;
constexpr int T_  = 4096;
constexpr int E_  = 2048;
constexpr int D_  = 1024;
constexpr int L_  = 2;
constexpr int M_  = 2;
constexpr int G_  = 4 * D_;     // 4096 gate rows per dir
constexpr int IN_ = 2048;       // layer input width (E == 2*D)

// ---------------------------------------------------------------------------
// CDNA5 types
// ---------------------------------------------------------------------------
typedef __attribute__((ext_vector_type(16))) __bf16 v16bf;
typedef __attribute__((ext_vector_type(8)))  float  v8f;
typedef __attribute__((ext_vector_type(4))) unsigned int tdm_g0_t;
typedef __attribute__((ext_vector_type(8))) int          tdm_g1_t;
typedef __attribute__((ext_vector_type(4))) int          tdm_g2_t;

union Frag16 { v16bf v; uint32_t u[8]; };

__device__ __forceinline__ uint16_t f2bf(float f) {
    uint32_t u = __float_as_uint(f);
    uint32_t r = u + 0x7FFFu + ((u >> 16) & 1u);   // round-to-nearest-even
    return (uint16_t)(r >> 16);
}

__device__ __forceinline__ float sigmoidf_(float x) {
    return 1.0f / (1.0f + expf(-x));
}

// ---------------------------------------------------------------------------
// K1: fp32 -> bf16 weight conversion (Wxh then Whh)
// ---------------------------------------------------------------------------
__global__ void convert_bf16_kernel(const float* __restrict__ wxh,
                                    const float* __restrict__ whh,
                                    uint16_t* __restrict__ wx_bf,
                                    uint16_t* __restrict__ whh_bf) {
    const long n1 = (long)L_ * 2 * G_ * IN_;
    const long n2 = (long)L_ * 2 * G_ * D_;
    for (long i = (long)blockIdx.x * blockDim.x + threadIdx.x; i < n1 + n2;
         i += (long)gridDim.x * blockDim.x) {
        if (i < n1) wx_bf[i] = f2bf(wxh[i]);
        else        whh_bf[i - n1] = f2bf(whh[i - n1]);
    }
}

// ---------------------------------------------------------------------------
// K2: embedding gather -> bf16 activations  x[b][t][e]
// ---------------------------------------------------------------------------
__global__ void embed_kernel(const int* __restrict__ tokens,
                             const float* __restrict__ emb,
                             uint16_t* __restrict__ x_bf) {
    const long N = (long)B_ * T_ * E_;
    for (long i = (long)blockIdx.x * blockDim.x + threadIdx.x; i < N;
         i += (long)gridDim.x * blockDim.x) {
        long bt = i >> 11;            // /E_
        int  e  = (int)(i & (E_ - 1));
        int  tok = tokens[bt];
        x_bf[i] = f2bf(emb[(size_t)tok * E_ + e]);
    }
}

// ---------------------------------------------------------------------------
// K3: gx[dir][t][g][b] = x(row) @ Wx^T + (bxh+bhh)     (bf16 WMMA GEMM)
//   M = B*T = 16384 rows, N = 4096 gates, K = 2048.
//   Workgroup tile 64(M) x 128(N); 8 waves in a 4x2 grid; K step 32.
//   A tile staged fragment-ready in LDS; B tile staged by the TDM.
// ---------------------------------------------------------------------------
__global__ __launch_bounds__(256)
void gx_gemm_kernel(const uint16_t* __restrict__ x_bf,
                    const uint16_t* __restrict__ wx_bf,
                    const float* __restrict__ bxh,
                    const float* __restrict__ bhh,
                    float* __restrict__ gx,
                    int layer) {
    __shared__ alignas(32) uint16_t As[4][32][16];   // fragment-ready A (4 KB)
    __shared__ alignas(32) uint16_t Bs[128][32];     // row-major weight tile (8 KB)

    const int dir  = blockIdx.z;
    const int n0   = blockIdx.x * 128;      // gate base
    const int m0   = blockIdx.y * 64;       // flattened (b,t) base
    const int tid  = threadIdx.x;
    const int wv   = tid >> 5;
    const int lane = tid & 31;
    const int wave_m = wv & 3;              // 4 row sub-tiles of 16
    const int wave_n = wv >> 2;             // 2 col groups of 64

    v8f acc[4];
#pragma unroll
    for (int i = 0; i < 4; ++i)
#pragma unroll
        for (int e = 0; e < 8; ++e) acc[i][e] = 0.0f;

    // --- A staging coordinates (one 16B chunk per thread per K-tile) ---
    const int r  = tid >> 2;                    // row 0..63 in tile
    const int c  = tid & 3;                     // 8-elem K chunk 0..3
    const int m  = m0 + r;
    const int bb = m >> 12;                     // /T_
    const int tt = m & (T_ - 1);
    const int trow = dir ? (T_ - 1 - tt) : tt;  // reversed sequence for dir R
    const size_t xrow = ((size_t)bb * T_ + trow) * IN_;
    const int sub   = r >> 4;
    const int alane = (r & 15) | ((c >> 1) << 4);  // lane = row%16 | half<<4
    const int ae0   = (c & 1) * 8;                 // frag element base

    const size_t wbase = (size_t)(layer * 2 + dir) * G_ * IN_ + (size_t)n0 * IN_;

#pragma unroll 1
    for (int k0 = 0; k0 < IN_; k0 += 32) {
        // A: global b128 -> fragment-ready LDS
        *(uint4*)&As[sub][alane][ae0] = *(const uint4*)&x_bf[xrow + k0 + c * 8];

        // B: 128x32 weight tile via Tensor Data Mover (wave 0 issues; EXEC-free DMA)
        if (wv == 0) {
            unsigned ldsOff = (unsigned)(size_t)&Bs[0][0];
            unsigned long long ga =
                (unsigned long long)(size_t)wx_bf + 2ull * (wbase + (size_t)k0);
            tdm_g0_t g0;
            g0[0] = 1u;                              // count=1 (valid D#)
            g0[1] = ldsOff;                          // lds_addr
            g0[2] = (unsigned)(ga & 0xffffffffu);    // global_addr lo
            g0[3] = (unsigned)((ga >> 32) & 0x1ffffffu) | (2u << 30); // hi | type=2
            tdm_g1_t g1;
            g1[0] = (int)(1u << 16);                             // data_size = 2B
            g1[1] = (int)((unsigned)(IN_ & 0xffff) << 16);       // tensor_dim0 lo
            g1[2] = (int)(((unsigned)IN_ >> 16) |
                          ((unsigned)(G_ & 0xffff) << 16));      // dim0 hi | dim1 lo
            g1[3] = (int)(((unsigned)G_ >> 16) | (32u << 16));   // dim1 hi | tile_dim0=32
            g1[4] = (int)128u;                                   // tile_dim1=128
            g1[5] = (int)IN_;                                    // dim0_stride lo (2048)
            g1[6] = 0;
            g1[7] = 0;
            tdm_g2_t g2;
            g2[0] = 0; g2[1] = 0; g2[2] = 0; g2[3] = 0;
            tdm_g1_t gz8;
            gz8[0] = 0; gz8[1] = 0; gz8[2] = 0; gz8[3] = 0;
            gz8[4] = 0; gz8[5] = 0; gz8[6] = 0; gz8[7] = 0;
            // 6-arg form (clang-23 / therock-10.0 headers): extra int32x8 group
            __builtin_amdgcn_tensor_load_to_lds(g0, g1, g2, g2, gz8, 0);
            __builtin_amdgcn_s_wait_tensorcnt(0);
        }
        __syncthreads();

        // A fragment: contiguous 32B per lane
        Frag16 fa;
        fa.v = *(const v16bf*)&As[wave_m][lane][0];

#pragma unroll
        for (int i = 0; i < 4; ++i) {
            // B fragment: lane = column (gate), half selects K 0-15/16-31; contiguous
            const uint16_t* bp = &Bs[wave_n * 64 + i * 16 + (lane & 15)][(lane >> 4) * 16];
            v16bf fb = *(const v16bf*)bp;
            acc[i] = __builtin_amdgcn_wmma_f32_16x16x32_bf16(
                false, fa.v, false, fb, (short)0, acc[i], false, false);
        }
        __syncthreads();
    }

    // --- store: gx layout [dir][t][g][B] (batch-minor for recurrent reads) ---
    const float* bx = bxh + (size_t)(layer * 2 + dir) * G_;
    const float* bh = bhh + (size_t)(layer * 2 + dir) * G_;
    const int nc    = lane & 15;
    const int rhalf = (lane >> 4) * 8;
#pragma unroll
    for (int i = 0; i < 4; ++i) {
        const int g = n0 + wave_n * 64 + i * 16 + nc;
        const float bias = bx[g] + bh[g];
#pragma unroll
        for (int e = 0; e < 8; ++e) {
            const int mrow = m0 + wave_m * 16 + rhalf + e;
            const int b2 = mrow >> 12;
            const int t2 = mrow & (T_ - 1);
            const size_t idx = ((((size_t)dir * T_ + t2) * G_ + g) << 2) + b2;
            gx[idx] = acc[i][e] + bias;
        }
    }
}

// ---------------------------------------------------------------------------
// K4: sequential LSTM recurrence, one workgroup per direction.
//   32 wave32 waves; wave w owns gate rows [w*128, w*128+128).
//   Per step: g = gx[t] + Whh @ h  (Whh streamed bf16 from L2, B=4 batches in
//   the WMMA N dimension), gates -> LDS, c/h update, h re-packed bf16 in LDS.
// ---------------------------------------------------------------------------
__global__ __launch_bounds__(1024)
void lstm_rec_kernel(const uint16_t* __restrict__ whh_bf,
                     const float* __restrict__ gx,
                     float* __restrict__ hL,
                     float* __restrict__ hR,
                     uint16_t* __restrict__ x_next,
                     int layer, int write_next) {
    __shared__ alignas(32) uint16_t hbuf[16][D_];   // rows 4..15 stay zero (32 KB)
    __shared__ float cbuf[B_][D_];                  // 16 KB
    __shared__ float gact[B_][G_];                  // 64 KB

    const int dir  = blockIdx.x;
    const int tid  = threadIdx.x;
    const int wv   = tid >> 5;
    const int lane = tid & 31;

    for (int i = tid; i < 16 * D_; i += 1024) (&hbuf[0][0])[i] = 0;
    for (int i = tid; i < B_ * D_; i += 1024) (&cbuf[0][0])[i] = 0.0f;
    __syncthreads();

    const uint16_t* W = whh_bf + (size_t)(layer * 2 + dir) * G_ * D_;
    float* hout = dir ? hR : hL;

    const int rbase = wv * 128;        // first gate row owned by this wave
    const int nc    = lane & 15;       // A row-in-subtile / C column (batch)
    const int half  = lane >> 4;

#pragma unroll 1
    for (int t = 0; t < T_; ++t) {
        // In a cluster dispatch this is the per-step h-exchange barrier;
        // outside a cluster the HW treats it as S_NOP (ISA 2. Cluster Barriers).
        __builtin_amdgcn_s_cluster_barrier();

        v8f acc[8];
#pragma unroll
        for (int s = 0; s < 8; ++s)
#pragma unroll
            for (int e = 0; e < 8; ++e) acc[s][e] = 0.0f;

#pragma unroll 1
        for (int ks = 0; ks < 32; ++ks) {
            // B fragment: h (bf16) from LDS; column = batch (4 live, 12 zero rows)
            v16bf fb = *(const v16bf*)&hbuf[nc][ks * 32 + half * 16];
#pragma unroll
            for (int s = 0; s < 8; ++s) {
                const int row = rbase + s * 16 + nc;
                const size_t base = (size_t)row * D_ + ks * 32 + half * 16;
                Frag16 fa;
                *(uint4*)&fa.u[0] = *(const uint4*)&W[base];       // K  0..7 (+half*16)
                *(uint4*)&fa.u[4] = *(const uint4*)&W[base + 8];   // K  8..15 (+half*16)
                acc[s] = __builtin_amdgcn_wmma_f32_16x16x32_bf16(
                    false, fa.v, false, fb, (short)0, acc[s], false, false);
            }
        }

        // add gx, apply gate nonlinearities, scatter to LDS (cols 0..3 = batches)
        const int bcol  = nc;
        const int rhalf = half * 8;
        if (bcol < B_) {
            const float* gxp = gx + ((((size_t)dir * T_ + t) * G_) << 2) + bcol;
#pragma unroll
            for (int s = 0; s < 8; ++s) {
#pragma unroll
                for (int e = 0; e < 8; ++e) {
                    const int gr = rbase + s * 16 + rhalf + e;
                    const float gsum = acc[s][e] + gxp[(size_t)gr << 2];
                    const int gate = gr >> 10;          // 0:i 1:g 2:f 3:o
                    gact[bcol][gr] = (gate == 1) ? tanhf(gsum) : sigmoidf_(gsum);
                }
            }
        }
        __syncthreads();

        // c/h update: thread j per hidden unit, all 4 batches
        {
            const int j = tid;
#pragma unroll
            for (int n = 0; n < B_; ++n) {
                const float iv = gact[n][j];
                const float gv = gact[n][D_ + j];
                const float fv = gact[n][2 * D_ + j];
                const float ov = gact[n][3 * D_ + j];
                const float cv = fv * cbuf[n][j] + iv * gv;
                cbuf[n][j] = cv;
                const float hv = ov * tanhf(cv);
                hbuf[n][j] = f2bf(hv);
                hout[((size_t)n * T_ + t) * D_ + j] = hv;
                if (write_next) {
                    const int torig = dir ? (T_ - 1 - t) : t;
                    x_next[((size_t)n * T_ + torig) * IN_ + dir * D_ + j] = f2bf(hv);
                }
            }
        }
        __syncthreads();
    }
}

// ---------------------------------------------------------------------------
// K5: attention pooling + output projection, one workgroup per batch
// ---------------------------------------------------------------------------
__global__ __launch_bounds__(1024)
void attn_out_kernel(const float* __restrict__ hL,
                     const float* __restrict__ hR,
                     const float* __restrict__ WcyL,
                     const float* __restrict__ WcyR,
                     float* __restrict__ out) {
    __shared__ float wsm[T_];              // scores / softmax weights (16 KB)
    __shared__ float lastL[D_], lastR[D_];
    __shared__ float red[32];
    __shared__ float stat;
    __shared__ float sacc[M_];

    const int b = blockIdx.x, tid = threadIdx.x, wv = tid >> 5, lane = tid & 31;

    lastL[tid] = hL[((size_t)b * T_ + (T_ - 1)) * D_ + tid];
    lastR[tid] = hR[((size_t)b * T_ + (T_ - 1)) * D_ + tid];
    if (tid < M_) sacc[tid] = 0.0f;
    __syncthreads();

    // scores: one wave per timestep, lane-split dot products
    for (int t = wv; t < T_; t += 32) {
        const float* pL = hL + ((size_t)b * T_ + t) * D_;
        const float* pR = hR + ((size_t)b * T_ + t) * D_;
        float s = 0.0f;
        for (int k = lane; k < D_; k += 32)
            s += pL[k] * lastL[k] + pR[k] * lastR[k];
        for (int off = 16; off; off >>= 1) s += __shfl_xor(s, off, 32);
        if (lane == 0) wsm[t] = s * (1.0f / 64.0f);   // 1/sqrt(T)
    }
    __syncthreads();

    // softmax: max
    float mx = -3.4e38f;
    for (int i = tid; i < T_; i += 1024) mx = fmaxf(mx, wsm[i]);
    for (int off = 16; off; off >>= 1) mx = fmaxf(mx, __shfl_xor(mx, off, 32));
    if (lane == 0) red[wv] = mx;
    __syncthreads();
    if (wv == 0) {
        float mm = red[lane];
        for (int off = 16; off; off >>= 1) mm = fmaxf(mm, __shfl_xor(mm, off, 32));
        if (lane == 0) stat = mm;
    }
    __syncthreads();
    const float smax = stat;

    // exp + sum
    float ps = 0.0f;
    for (int i = tid; i < T_; i += 1024) {
        float e = expf(wsm[i] - smax);
        wsm[i] = e;
        ps += e;
    }
    for (int off = 16; off; off >>= 1) ps += __shfl_xor(ps, off, 32);
    if (lane == 0) red[wv] = ps;
    __syncthreads();
    if (wv == 0) {
        float ss = red[lane];
        for (int off = 16; off; off >>= 1) ss += __shfl_xor(ss, off, 32);
        if (lane == 0) stat = ss;
    }
    __syncthreads();
    const float inv = 1.0f / stat;

    // context vectors: thread j accumulates over t (coalesced per t)
    const int j = tid;
    float aL = 0.0f, aR = 0.0f;
    for (int t = 0; t < T_; ++t) {
        const float a = wsm[t];
        aL += a * hL[((size_t)b * T_ + t) * D_ + j];
        aR += a * hR[((size_t)b * T_ + t) * D_ + j];
    }
    aL *= inv;
    aR *= inv;

    // output projection via wave reduce + LDS float atomics (ds_add_f32)
#pragma unroll
    for (int mi = 0; mi < M_; ++mi) {
        float p = WcyL[mi * D_ + j] * aL + WcyR[mi * D_ + j] * aR;
        for (int off = 16; off; off >>= 1) p += __shfl_xor(p, off, 32);
        if (lane == 0) atomicAdd(&sacc[mi], p);
    }
    __syncthreads();
    if (tid < M_) out[b * M_ + tid] = sacc[tid];
}

// ---------------------------------------------------------------------------
// Host launcher
// ---------------------------------------------------------------------------
extern "C" void kernel_launch(void* const* d_in, const int* in_sizes, int n_in,
                              void* d_out, int out_size, void* d_ws, size_t ws_size,
                              hipStream_t stream) {
    const int*   tokens = (const int*)d_in[0];
    const float* emb    = (const float*)d_in[1];
    const float* Wxh    = (const float*)d_in[2];
    const float* Whh    = (const float*)d_in[3];
    const float* bxh    = (const float*)d_in[4];
    const float* bhh    = (const float*)d_in[5];
    const float* WcyL   = (const float*)d_in[6];
    const float* WcyR   = (const float*)d_in[7];
    float* out = (float*)d_out;

    // Workspace carve-up (bytes)
    char* ws = (char*)d_ws;
    const size_t sz_wx  = (size_t)L_ * 2 * G_ * IN_ * 2;   //  64 MB bf16 Wxh
    const size_t sz_whh = (size_t)L_ * 2 * G_ * D_ * 2;    //  32 MB bf16 Whh
    const size_t sz_x   = (size_t)B_ * T_ * IN_ * 2;       //  64 MB bf16 activations
    const size_t sz_gx  = (size_t)2 * T_ * G_ * B_ * 4;    // 512 MB fp32 gx
    const size_t sz_h   = (size_t)B_ * T_ * D_ * 4;        //  64 MB fp32 each

    uint16_t* wx_bf  = (uint16_t*)(ws);
    uint16_t* whh_bf = (uint16_t*)(ws + sz_wx);
    uint16_t* x_bf   = (uint16_t*)(ws + sz_wx + sz_whh);
    float*    gxp    = (float*)(ws + sz_wx + sz_whh + sz_x);
    float*    hLp    = (float*)(ws + sz_wx + sz_whh + sz_x + sz_gx);
    float*    hRp    = (float*)(ws + sz_wx + sz_whh + sz_x + sz_gx + sz_h);

    convert_bf16_kernel<<<2048, 256, 0, stream>>>(Wxh, Whh, wx_bf, whh_bf);
    embed_kernel<<<4096, 256, 0, stream>>>(tokens, emb, x_bf);

    for (int layer = 0; layer < L_; ++layer) {
        gx_gemm_kernel<<<dim3(G_ / 128, (B_ * T_) / 64, 2), 256, 0, stream>>>(
            x_bf, wx_bf, bxh, bhh, gxp, layer);
        lstm_rec_kernel<<<2, 1024, 0, stream>>>(
            whh_bf, gxp, hLp, hRp, x_bf, layer, (layer < L_ - 1) ? 1 : 0);
    }

    attn_out_kernel<<<B_, 1024, 0, stream>>>(hLp, hRp, WcyL, WcyR, out);
}
